// RealAgnosticAttResidualInteractionBlock_18502719111825
// MI455X (gfx1250) — matrix operand order repaired and checked by reading, not compiled, not run on hardware
//
#include <hip/hip_runtime.h>
#include <hip/hip_fp16.h>

// ---------------- problem constants ----------------
constexpr int   kNodes    = 5000;
constexpr int   kEdges    = 80000;
constexpr float kInvSqrt3 = 0.57735026918962576f;
constexpr float kInvF     = 0.08838834764831845f;   // 1/sqrt(128)
constexpr float kS0       = 0.08574929257125442f;   // 1/sqrt(136)
constexpr float kS1       = 0.0625f;                // 1/sqrt(256)
constexpr float kInvM     = 1.0f / 256.0f;          // 1/(sqrt(256)*16)
constexpr int   kKp0      = 160;                    // 136 padded to mult of 32

// LDS pitches: row stride mod 64 banks = 20 / 4 dwords (conflict-free b128 fragment
// loads) while keeping 16B alignment: 168 / 264 halves.
constexpr int kPAug = 168;
constexpr int kPH   = 264;

// ---------------- workspace layout (bytes) ----------------
constexpr size_t OFF_W0T = 0;                              // mlp_w0^T  256x160 f16
constexpr size_t OFF_W1T = OFF_W0T + 256 * 160 * 2;        // mlp_w1^T  256x256
constexpr size_t OFF_W2T = OFF_W1T + 256 * 256 * 2;        // mlp_w2^T  256x256
constexpr size_t OFF_W3T = OFF_W2T + 256 * 256 * 2;        // mlp_w3^T  512x256
constexpr size_t OFF_L0T = OFF_W3T + 512 * 256 * 2;        // W_lin0^T  128x256
constexpr size_t OFF_L1T = OFF_L0T + 128 * 256 * 2;        // W_lin1^T  128x256
constexpr size_t OFF_U0T = OFF_L1T + 128 * 256 * 2;        // W_up0^T   128x128
constexpr size_t OFF_U1T = OFF_U0T + 128 * 128 * 2;        // W_up1^T   128x128
constexpr size_t OFF_DT  = OFF_U1T + 128 * 128 * 2;        // W_down^T   64x128
constexpr size_t OFF_K0T = OFF_DT  +  64 * 128 * 2;        // W_skip0^T 128x128
constexpr size_t OFF_K1T = OFF_K0T + 128 * 128 * 2;        // W_skip1^T 128x128
constexpr size_t OFF_SUP = OFF_K1T + 128 * 128 * 2;        // s_up  5000x128 f32
constexpr size_t OFF_VUP = OFF_SUP + (size_t)kNodes * 128 * 4;     // v_up 3x5000x128
constexpr size_t OFF_DWN = OFF_VUP + (size_t)3 * kNodes * 128 * 4; // down 5000x64
constexpr size_t OFF_MS  = OFF_DWN + (size_t)kNodes * 64 * 4;      // ms 5000x256
constexpr size_t OFF_MV  = OFF_MS  + (size_t)kNodes * 256 * 4;     // mv 3x5000x256

typedef __attribute__((ext_vector_type(4)))  float    v4f;
typedef __attribute__((ext_vector_type(8)))  float    v8f;
typedef __attribute__((ext_vector_type(8)))  _Float16 v8h;
typedef __attribute__((ext_vector_type(16))) _Float16 v16h;

__device__ __forceinline__ v8f wmma16(v16h a, v16h b, v8f c) {
  return __builtin_amdgcn_wmma_f32_16x16x32_f16(false, a, false, b, (short)0, c,
                                                false, false);
}
__device__ __forceinline__ void atomAddF(float* p, float v) {
  (void)__hip_atomic_fetch_add(p, v, __ATOMIC_RELAXED, __HIP_MEMORY_SCOPE_AGENT);
}
__device__ __forceinline__ v16h cat16(v8h lo, v8h hi) {
  return __builtin_shufflevector(lo, hi, 0, 1, 2, 3, 4, 5, 6, 7,
                                 8, 9, 10, 11, 12, 13, 14, 15);
}

// ---------------- weight prep: f32 (K,N) -> f16 transposed (N,Kp), zero-pad K ----
__global__ void k_transpose_f16(const float* __restrict__ W, _Float16* __restrict__ WT,
                                int K, int N, int Kp) {
  int idx = blockIdx.x * blockDim.x + threadIdx.x;
  if (idx >= N * Kp) return;
  int n = idx / Kp, k = idx - n * Kp;
  WT[idx] = (_Float16)((k < K) ? W[k * N + n] : 0.0f);
}

__global__ void k_zero_f32(float* __restrict__ p, int n) {
  int i = blockIdx.x * blockDim.x + threadIdx.x;
  int stride = gridDim.x * blockDim.x;
  for (; i < n; i += stride) p[i] = 0.0f;
}

// ---------------- generic WMMA GEMM ----------------
// C[m*ldc + n*sC] = scale * sum_k A[m*lda + k*sA] * BT[n*Kp + k]
// BT: f16 (N x Kp), K zero-padded to Kp (mult of 32, <= 256).
__global__ __launch_bounds__(128) void k_gemm_tn(
    const float* __restrict__ A, int lda, int sA,
    const _Float16* __restrict__ BT, int Kp,
    float* __restrict__ C, int ldc, int sC,
    int M, int K, int N, float scale) {
  __shared__ _Float16 sA16[16][kPH];     // 16 x 264 halves (Kp <= 256)
  const int tid   = threadIdx.x;
  const int lane  = tid & 31;
  const int wave  = tid >> 5;
  const int sh    = lane >> 4;
  const int mbase = blockIdx.x * 16;

  for (int idx = tid; idx < 16 * Kp; idx += 128) {
    int r = idx / Kp, k = idx - r * Kp;
    int m = mbase + r;
    float x = (m < M && k < K) ? A[m * lda + k * sA] : 0.0f;
    sA16[r][k] = (_Float16)x;
  }
  __syncthreads();

  const _Float16* arow = &sA16[lane & 15][0];
  for (int nt = wave; nt < (N >> 4); nt += 4) {
    const int ncol = nt * 16 + (lane & 15);
    const _Float16* brow = BT + ncol * Kp;
    v8f acc = {};
    for (int kk = 0; kk < Kp; kk += 32) {
      v8h alo = *(const v8h*)(arow + kk + sh * 16);
      v8h ahi = *(const v8h*)(arow + kk + sh * 16 + 8);
      v16h b  = *(const v16h*)(brow + kk + sh * 16);
      acc = wmma16(cat16(alo, ahi), b, acc);
    }
#pragma unroll
    for (int r = 0; r < 8; ++r) {
      int m = mbase + sh * 8 + r;
      if (m < M) C[m * ldc + ncol * sC] = acc[r] * scale;
    }
  }
}

// ---------------- fused edge kernel -----------------------------------------
// 32 edges / block (two 16-row strips sharing every B fragment), MLP via WMMA,
// layer-4 tensor-product weights scattered straight from accumulator registers.
__global__ __launch_bounds__(128) void k_edge_mlp_tp(
    const float* __restrict__ edge_attrs,   // (E,4)
    const float* __restrict__ edge_feats,   // (E,8)
    const int*   __restrict__ edge_index,   // (2,E)
    const float* __restrict__ down,         // (N,64)
    const float* __restrict__ s_up,         // (N,128)
    const float* __restrict__ v_up,         // (3,N,128)
    const _Float16* __restrict__ w0T,       // (256,160)
    const _Float16* __restrict__ w1T,       // (256,256)
    const _Float16* __restrict__ w2T,       // (256,256)
    const _Float16* __restrict__ w3T,       // (512,256)
    float* __restrict__ ms,                 // (N,256)
    float* __restrict__ mv)                 // (3,N,256)
{
  __shared__ _Float16 sAug[32][kPAug];      // ~10.5 KB
  __shared__ _Float16 sH[2][32][kPH];       // ~33.8 KB ping-pong
  __shared__ int   sSnd[32], sRcv[32];
  __shared__ float sY0[32], sY1x[32], sY1y[32], sY1z[32];
  const int tid  = threadIdx.x;
  const int lane = tid & 31;
  const int wave = tid >> 5;
  const int sh   = lane >> 4;
  const int eb   = blockIdx.x * 32;

  // ---- per-edge metadata into LDS ----
  if (tid < 32) {
    int ge = eb + tid;
    sSnd[tid] = edge_index[ge];
    sRcv[tid] = edge_index[kEdges + ge];
    sY0[tid]  = edge_attrs[ge * 4 + 0];
    sY1x[tid] = edge_attrs[ge * 4 + 1];
    sY1y[tid] = edge_attrs[ge * 4 + 2];
    sY1z[tid] = edge_attrs[ge * 4 + 3];
  }
  // ---- stage aug = [edge_feats | down[snd] | down[rcv] | 0-pad] as f16 ----
  for (int it = tid; it < 256; it += 128) {
    const int e = it >> 3, prt = it & 7, ge = eb + e;
    const int snd = edge_index[ge];
    const int rcv = edge_index[kEdges + ge];
    v4f a = *(const v4f*)(down + snd * 64 + prt * 8);
    v4f b = *(const v4f*)(down + snd * 64 + prt * 8 + 4);
    v8h h;
#pragma unroll
    for (int q = 0; q < 4; ++q) { h[q] = (_Float16)a[q]; h[4 + q] = (_Float16)b[q]; }
    *(v8h*)(&sAug[e][8 + prt * 8]) = h;
    a = *(const v4f*)(down + rcv * 64 + prt * 8);
    b = *(const v4f*)(down + rcv * 64 + prt * 8 + 4);
#pragma unroll
    for (int q = 0; q < 4; ++q) { h[q] = (_Float16)a[q]; h[4 + q] = (_Float16)b[q]; }
    *(v8h*)(&sAug[e][72 + prt * 8]) = h;
    if (prt == 0) {
      a = *(const v4f*)(edge_feats + ge * 8);
      b = *(const v4f*)(edge_feats + ge * 8 + 4);
#pragma unroll
      for (int q = 0; q < 4; ++q) { h[q] = (_Float16)a[q]; h[4 + q] = (_Float16)b[q]; }
      *(v8h*)(&sAug[e][0]) = h;
      v8h z = {};
      *(v8h*)(&sAug[e][136]) = z;
      *(v8h*)(&sAug[e][144]) = z;
      *(v8h*)(&sAug[e][152]) = z;
    }
  }
  // prefetch hot MLP weights (global_prefetch_b8) while aug settles
  {
    const char* wbase = (const char*)w0T;
    for (int off = tid * 256; off < 212992; off += 128 * 256)
      __builtin_prefetch(wbase + off, 0, 3);
  }
  __syncthreads();

  // two 16-row strips share each B fragment: 1 v16h global load -> 2 WMMAs
  auto runL = [&](const _Float16* src, int sp, const _Float16* BT, int Kp,
                  int N, float scale, _Float16* dst, int dp) {
    const _Float16* arow0 = src + (lane & 15) * sp;
    const _Float16* arow1 = src + (16 + (lane & 15)) * sp;
    for (int nt = wave; nt < (N >> 4); nt += 4) {
      const int ncol = nt * 16 + (lane & 15);
      const _Float16* brow = BT + ncol * Kp;
      v8f acc0 = {}, acc1 = {};
      for (int kk = 0; kk < Kp; kk += 32) {
        v16h b  = *(const v16h*)(brow + kk + sh * 16);
        v16h a0 = cat16(*(const v8h*)(arow0 + kk + sh * 16),
                        *(const v8h*)(arow0 + kk + sh * 16 + 8));
        v16h a1 = cat16(*(const v8h*)(arow1 + kk + sh * 16),
                        *(const v8h*)(arow1 + kk + sh * 16 + 8));
        acc0 = wmma16(a0, b, acc0);
        acc1 = wmma16(a1, b, acc1);
      }
#pragma unroll
      for (int r = 0; r < 8; ++r) {
        float x = acc0[r] * scale;
        x = x / (1.0f + __expf(-x));                       // SiLU
        dst[(sh * 8 + r) * dp + ncol] = (_Float16)x;
        x = acc1[r] * scale;
        x = x / (1.0f + __expf(-x));
        dst[(16 + sh * 8 + r) * dp + ncol] = (_Float16)x;
      }
    }
  };

  runL(&sAug[0][0], kPAug, w0T, kKp0, 256, kS0, &sH[0][0][0], kPH);
  __syncthreads();
  runL(&sH[0][0][0], kPH, w1T, 256, 256, kS1, &sH[1][0][0], kPH);
  __syncthreads();
  runL(&sH[1][0][0], kPH, w2T, 256, 256, kS1, &sH[0][0][0], kPH);
  __syncthreads();

  // ---- layer 4 + tensor product + scatter, straight from accumulators ----
  // tp columns: wA = tiles 0..7, wB = 8..15, wC = 16..23, wD = 24..31.
  // Lane (sh,r,n) owns (edge = strip*16 + sh*8 + r, channel = g*16 + n):
  // gathers and atomics are coalesced across the 16 lanes of each half-wave.
  const int plane = kNodes * 128;
  float* mv0 = mv;
  float* mv1 = mv + kNodes * 256;
  float* mv2 = mv + 2 * kNodes * 256;
  for (int g = wave; g < 8; g += 4) {
    const int n15 = lane & 15;
    const int col = g * 16 + n15;                 // channel 0..127
    for (int s = 0; s < 2; ++s) {
      const _Float16* arow = &sH[0][s * 16 + n15][0];
      // -- wA / wB pair (shares s_up gather) --
      {
        const _Float16* browA = w3T + (g * 16 + n15) * 256;
        const _Float16* browB = w3T + ((g + 8) * 16 + n15) * 256;
        v8f accA = {}, accB = {};
        for (int kk = 0; kk < 256; kk += 32) {
          v16h a = cat16(*(const v8h*)(arow + kk + sh * 16),
                         *(const v8h*)(arow + kk + sh * 16 + 8));
          v16h bA = *(const v16h*)(browA + kk + sh * 16);
          v16h bB = *(const v16h*)(browB + kk + sh * 16);
          accA = wmma16(a, bA, accA);
          accB = wmma16(a, bB, accB);
        }
#pragma unroll
        for (int r = 0; r < 8; ++r) {
          int e = s * 16 + sh * 8 + r;
          int snd = sSnd[e], rcv = sRcv[e];
          float xs = s_up[snd * 128 + col];
          atomAddF(ms + rcv * 256 + col, accA[r] * kS1 * xs * sY0[e]);
          float t = accB[r] * kS1 * xs * kInvSqrt3;
          atomAddF(mv0 + rcv * 256 + col, t * sY1x[e]);
          atomAddF(mv1 + rcv * 256 + col, t * sY1y[e]);
          atomAddF(mv2 + rcv * 256 + col, t * sY1z[e]);
        }
      }
      // -- wC / wD pair (shares v_up gathers) --
      {
        const _Float16* browC = w3T + ((g + 16) * 16 + n15) * 256;
        const _Float16* browD = w3T + ((g + 24) * 16 + n15) * 256;
        v8f accC = {}, accD = {};
        for (int kk = 0; kk < 256; kk += 32) {
          v16h a = cat16(*(const v8h*)(arow + kk + sh * 16),
                         *(const v8h*)(arow + kk + sh * 16 + 8));
          v16h bC = *(const v16h*)(browC + kk + sh * 16);
          v16h bD = *(const v16h*)(browD + kk + sh * 16);
          accC = wmma16(a, bC, accC);
          accD = wmma16(a, bD, accD);
        }
#pragma unroll
        for (int r = 0; r < 8; ++r) {
          int e = s * 16 + sh * 8 + r;
          int snd = sSnd[e], rcv = sRcv[e];
          float x0 = v_up[snd * 128 + col];
          float x1 = v_up[plane + snd * 128 + col];
          float x2 = v_up[2 * plane + snd * 128 + col];
          int c = 128 + col;
          float t = accC[r] * kS1 * sY0[e] * kInvSqrt3;
          atomAddF(mv0 + rcv * 256 + c, t * x0);
          atomAddF(mv1 + rcv * 256 + c, t * x1);
          atomAddF(mv2 + rcv * 256 + c, t * x2);
          atomAddF(ms + rcv * 256 + c,
                   accD[r] * kS1 *
                   (x0 * sY1x[e] + x1 * sY1y[e] + x2 * sY1z[e]) * kInvSqrt3);
        }
      }
    }
  }
}

// ---------------- launch ----------------
extern "C" void kernel_launch(void* const* d_in, const int* in_sizes, int n_in,
                              void* d_out, int out_size, void* d_ws, size_t ws_size,
                              hipStream_t stream) {
  (void)in_sizes; (void)n_in; (void)out_size; (void)ws_size;
  const float* node_feats = (const float*)d_in[1];
  const float* edge_attrs = (const float*)d_in[2];
  const float* edge_feats = (const float*)d_in[3];
  const int*   edge_index = (const int*)d_in[4];
  const float* W_up0   = (const float*)d_in[5];
  const float* W_up1   = (const float*)d_in[6];
  const float* W_down  = (const float*)d_in[7];
  const float* mlp_w0  = (const float*)d_in[8];
  const float* mlp_w1  = (const float*)d_in[9];
  const float* mlp_w2  = (const float*)d_in[10];
  const float* mlp_w3  = (const float*)d_in[11];
  const float* W_lin0  = (const float*)d_in[12];
  const float* W_lin1  = (const float*)d_in[13];
  const float* W_skip0 = (const float*)d_in[14];
  const float* W_skip1 = (const float*)d_in[15];

  char* ws = (char*)d_ws;
  _Float16* w0T = (_Float16*)(ws + OFF_W0T);
  _Float16* w1T = (_Float16*)(ws + OFF_W1T);
  _Float16* w2T = (_Float16*)(ws + OFF_W2T);
  _Float16* w3T = (_Float16*)(ws + OFF_W3T);
  _Float16* l0T = (_Float16*)(ws + OFF_L0T);
  _Float16* l1T = (_Float16*)(ws + OFF_L1T);
  _Float16* u0T = (_Float16*)(ws + OFF_U0T);
  _Float16* u1T = (_Float16*)(ws + OFF_U1T);
  _Float16* dT  = (_Float16*)(ws + OFF_DT);
  _Float16* k0T = (_Float16*)(ws + OFF_K0T);
  _Float16* k1T = (_Float16*)(ws + OFF_K1T);
  float* s_up = (float*)(ws + OFF_SUP);
  float* v_up = (float*)(ws + OFF_VUP);
  float* down = (float*)(ws + OFF_DWN);
  float* ms   = (float*)(ws + OFF_MS);
  float* mv   = (float*)(ws + OFF_MV);

  auto T = [&](const float* W, _Float16* WT, int K, int N, int Kp) {
    int total = N * Kp;
    k_transpose_f16<<<(total + 255) / 256, 256, 0, stream>>>(W, WT, K, N, Kp);
  };
  T(mlp_w0, w0T, 136, 256, 160);
  T(mlp_w1, w1T, 256, 256, 256);
  T(mlp_w2, w2T, 256, 256, 256);
  T(mlp_w3, w3T, 256, 512, 256);
  T(W_lin0, l0T, 256, 128, 256);
  T(W_lin1, l1T, 256, 128, 256);
  T(W_up0,  u0T, 128, 128, 128);
  T(W_up1,  u1T, 128, 128, 128);
  T(W_down, dT,  128,  64, 128);
  T(W_skip0, k0T, 128, 128, 128);
  T(W_skip1, k1T, 128, 128, 128);
  k_zero_f32<<<2048, 256, 0, stream>>>(ms, kNodes * 256 * 4);  // ms + mv contiguous

  float* out = (float*)d_out;                       // msg (5000,128,4)
  float* sc  = out + (size_t)kNodes * 512;          // sc  (5000,512)
  const int mb = (kNodes + 15) / 16;                // 313 row-tiles

  // ---- node GEMMs (K = 128) ----
  k_gemm_tn<<<mb, 128, 0, stream>>>(node_feats, 512, 1, k0T, 128, sc, 512, 1,
                                    kNodes, 128, 128, kInvF);            // sc_s
  for (int i = 0; i < 3; ++i)                                            // sc_v
    k_gemm_tn<<<mb, 128, 0, stream>>>(node_feats + 128 + i, 512, 3, k1T, 128,
                                      sc + 128 + i, 512, 3, kNodes, 128, 128, kInvF);
  k_gemm_tn<<<mb, 128, 0, stream>>>(node_feats, 512, 1, u0T, 128, s_up, 128, 1,
                                    kNodes, 128, 128, kInvF);            // s_up
  for (int i = 0; i < 3; ++i)                                            // v_up
    k_gemm_tn<<<mb, 128, 0, stream>>>(node_feats + 128 + i, 512, 3, u1T, 128,
                                      v_up + i * kNodes * 128, 128, 1,
                                      kNodes, 128, 128, kInvF);
  k_gemm_tn<<<mb, 128, 0, stream>>>(node_feats, 512, 1, dT, 128, down, 64, 1,
                                    kNodes, 128, 64, kInvF);             // down

  // ---- fused edge MLP + tensor product + scatter (32 edges / block) ----
  k_edge_mlp_tp<<<kEdges / 32, 128, 0, stream>>>(edge_attrs, edge_feats, edge_index,
                                                 down, s_up, v_up,
                                                 w0T, w1T, w2T, w3T, ms, mv);

  // ---- output GEMMs (K = 256) into strided msg layout ----
  k_gemm_tn<<<mb, 128, 0, stream>>>(ms, 256, 1, l0T, 256, out, 512, 4,
                                    kNodes, 256, 128, kInvM);            // out_s (j=0)
  for (int i = 0; i < 3; ++i)                                            // out_v (j=1..3)
    k_gemm_tn<<<mb, 128, 0, stream>>>(mv + i * kNodes * 256, 256, 1, l1T, 256,
                                      out + 1 + i, 512, 4, kNodes, 256, 128, kInvM);
}
